// AGNNNet_20847771254905
// MI455X (gfx1250) — matrix operand order repaired and checked by reading, not compiled, not run on hardware
//
#include <hip/hip_runtime.h>
#include <hip/hip_bf16.h>

typedef __attribute__((ext_vector_type(16))) _Float16 v16h;
typedef __attribute__((ext_vector_type(8)))  float    v8f;
typedef __attribute__((ext_vector_type(8)))  unsigned v8u;

#define N_FEAT 512
#define HID    16
#define N_CLS  10

__device__ __forceinline__ void atomicAddF(float* p, float v) {
    __hip_atomic_fetch_add(p, v, __ATOMIC_RELAXED, __HIP_MEMORY_SCOPE_AGENT);
}

// ---------------------------------------------------------------------------
// Pack W1 [512][16] f32 row-major into f16 WMMA B-fragment layout.
// For k-step s (K base = 32*s): lane l supplies column N = l&15,
//   K range (l<16 ? 0..15 : 16..31); dword d holds K = {koff+2d, koff+2d+1}.
// Stored at wpack[(s*32 + l)*8 + d] -> main loop loads 32 contiguous bytes.
// ---------------------------------------------------------------------------
__global__ void agnn_pack_w1(const float* __restrict__ W1, unsigned* __restrict__ wpack) {
    int l = threadIdx.x;             // 0..31
    int col  = l & 15;
    int koff = (l < 16) ? 0 : 16;
    for (int s = 0; s < 16; ++s) {
        int kb = s * 32 + koff;
        for (int d = 0; d < 8; ++d) {
            _Float16 h0 = (_Float16)W1[(kb + 2 * d)     * HID + col];
            _Float16 h1 = (_Float16)W1[(kb + 2 * d + 1) * HID + col];
            union { _Float16 h[2]; unsigned u; } pk;
            pk.h[0] = h0; pk.h[1] = h1;
            wpack[(s * 32 + l) * 8 + d] = pk.u;
        }
    }
}

// ---------------------------------------------------------------------------
// h = relu(x @ W1 + b1) via v_wmma_f32_16x16x32_f16. One wave per 16-row tile.
// A operand as loaded here (contiguous k, same order as wpack): lane<16 holds row=lane, K {0..15};
//                          lane>=16 holds row=lane-16, K {16..31}.
// ---------------------------------------------------------------------------
__global__ __launch_bounds__(256) void agnn_gemm1(
    const float* __restrict__ x, const unsigned* __restrict__ wpack,
    const float* __restrict__ b1, float* __restrict__ h, int nTiles) {
    int wave = threadIdx.x >> 5;
    int lane = threadIdx.x & 31;
    int tile = blockIdx.x * 8 + wave;
    if (tile >= nTiles) return;                   // uniform per-wave: EXEC stays full

    int row  = tile * 16 + (lane & 15);
    int koff = (lane < 16) ? 0 : 16;
    const float* xr = x + (size_t)row * N_FEAT;

    v8f c = {};
    for (int s = 0; s < 16; ++s) {
        const float* pa = xr + s * 32 + koff;
        float4 f0 = *(const float4*)(pa);
        float4 f1 = *(const float4*)(pa + 4);
        float4 f2 = *(const float4*)(pa + 8);
        float4 f3 = *(const float4*)(pa + 12);
        v16h a;
        a[0]  = (_Float16)f0.x; a[1]  = (_Float16)f0.y; a[2]  = (_Float16)f0.z; a[3]  = (_Float16)f0.w;
        a[4]  = (_Float16)f1.x; a[5]  = (_Float16)f1.y; a[6]  = (_Float16)f1.z; a[7]  = (_Float16)f1.w;
        a[8]  = (_Float16)f2.x; a[9]  = (_Float16)f2.y; a[10] = (_Float16)f2.z; a[11] = (_Float16)f2.w;
        a[12] = (_Float16)f3.x; a[13] = (_Float16)f3.y; a[14] = (_Float16)f3.z; a[15] = (_Float16)f3.w;

        v8u braw = *(const v8u*)(wpack + (size_t)(s * 32 + lane) * 8);
        v16h b = __builtin_bit_cast(v16h, braw);

        c = __builtin_amdgcn_wmma_f32_16x16x32_f16(
                false, a, false, b, (short)0, c, false, false);
    }

    // C layout: lane l, vgpr r -> (M = r + 8*(l>=16), N = l&15)
    int col = lane & 15;
    int rb  = (lane >> 4) * 8;
    float bias = b1[col];
    for (int r = 0; r < 8; ++r) {
        float v = c[r] + bias;
        v = v > 0.0f ? v : 0.0f;
        h[(size_t)(tile * 16 + rb + r) * HID + col] = v;
    }
}

// hn = h / max(||h||, eps); also store the clamped norm (h == hn * norm exactly)
__global__ void agnn_normalize(const float* __restrict__ h, float* __restrict__ hn,
                               float* __restrict__ nrm, int N) {
    int n = blockIdx.x * blockDim.x + threadIdx.x;
    if (n >= N) return;
    const float4* hp = (const float4*)(h + (size_t)n * HID);
    float4 v0 = hp[0], v1 = hp[1], v2 = hp[2], v3 = hp[3];
    float ss = v0.x*v0.x + v0.y*v0.y + v0.z*v0.z + v0.w*v0.w
             + v1.x*v1.x + v1.y*v1.y + v1.z*v1.z + v1.w*v1.w
             + v2.x*v2.x + v2.y*v2.y + v2.z*v2.z + v2.w*v2.w
             + v3.x*v3.x + v3.y*v3.y + v3.z*v3.z + v3.w*v3.w;
    float norm = fmaxf(sqrtf(ss), 1e-12f);
    float inv = 1.0f / norm;
    float4* op = (float4*)(hn + (size_t)n * HID);
    v0.x*=inv; v0.y*=inv; v0.z*=inv; v0.w*=inv;
    v1.x*=inv; v1.y*=inv; v1.z*=inv; v1.w*=inv;
    v2.x*=inv; v2.y*=inv; v2.z*=inv; v2.w*=inv;
    v3.x*=inv; v3.y*=inv; v3.z*=inv; v3.w*=inv;
    op[0]=v0; op[1]=v1; op[2]=v2; op[3]=v3;
    nrm[n] = norm;
}

__global__ void agnn_zero(float* __restrict__ p, int n) {
    int i = blockIdx.x * blockDim.x + threadIdx.x;
    if (i < n) p[i] = 0.0f;
}

// ---------------------------------------------------------------------------
// Edge pass: one edge per 16-lane group, lane j owns feature j.
// alpha = beta*dot(hn[s],hn[d]); since |alpha| <= |beta|, softmax shift = |beta|
// (exact by shift-invariance; replaces the reference's segment_max pass).
// acc[d] += exp(alpha-|beta|) * h[s];  denom[d] += exp(alpha-|beta|).
// Edges e >= E are the appended self-loops.
// ---------------------------------------------------------------------------
__global__ __launch_bounds__(256) void agnn_edges(
    const int* __restrict__ srcA, const int* __restrict__ dstA, int E, int Etot,
    const float* __restrict__ hn, const float* __restrict__ nrm,
    float* __restrict__ acc, float* __restrict__ denom,
    const float* __restrict__ betaPtr, int useBeta) {
    int gid = blockIdx.x * blockDim.x + threadIdx.x;
    int e = gid >> 4;
    int j = gid & 15;
    if (e >= Etot) return;
    int s, d;
    if (e < E) { s = srcA[e]; d = dstA[e]; }
    else       { s = e - E;   d = s; }

    float beta = useBeta ? betaPtr[0] : 1.0f;
    float hs = hn[(size_t)s * HID + j];
    float hd = hn[(size_t)d * HID + j];
    float p = hs * hd;
    p += __shfl_xor(p, 1);
    p += __shfl_xor(p, 2);
    p += __shfl_xor(p, 4);
    p += __shfl_xor(p, 8);                       // full 16-lane dot product
    float ea = __expf(beta * p - fabsf(beta));   // bounded in [e^{-2|b|}, 1]
    float contrib = ea * hs * nrm[s];            // == ea * h[s][j] exactly
    atomicAddF(&acc[(size_t)d * HID + j], contrib);
    if (j == 0) atomicAddF(&denom[d], ea);
}

__global__ void agnn_divide(const float* __restrict__ acc, const float* __restrict__ denom,
                            float* __restrict__ h, int n16) {
    int i = blockIdx.x * blockDim.x + threadIdx.x;
    if (i >= n16) return;
    h[i] = acc[i] / denom[i >> 4];
}

// logits = h @ W2 + b2 -> log_softmax
__global__ void agnn_head(const float* __restrict__ h, const float* __restrict__ W2,
                          const float* __restrict__ b2, float* __restrict__ out, int N) {
    int n = blockIdx.x * blockDim.x + threadIdx.x;
    if (n >= N) return;
    float hr[HID];
    const float4* hp = (const float4*)(h + (size_t)n * HID);
    float4 v0 = hp[0], v1 = hp[1], v2 = hp[2], v3 = hp[3];
    hr[0]=v0.x; hr[1]=v0.y; hr[2]=v0.z; hr[3]=v0.w;
    hr[4]=v1.x; hr[5]=v1.y; hr[6]=v1.z; hr[7]=v1.w;
    hr[8]=v2.x; hr[9]=v2.y; hr[10]=v2.z; hr[11]=v2.w;
    hr[12]=v3.x; hr[13]=v3.y; hr[14]=v3.z; hr[15]=v3.w;
    float z[N_CLS];
    #pragma unroll
    for (int c = 0; c < N_CLS; ++c) {
        float a = b2[c];
        #pragma unroll
        for (int j = 0; j < HID; ++j) a += hr[j] * W2[j * N_CLS + c];
        z[c] = a;
    }
    float m = z[0];
    #pragma unroll
    for (int c = 1; c < N_CLS; ++c) m = fmaxf(m, z[c]);
    float sum = 0.0f;
    #pragma unroll
    for (int c = 0; c < N_CLS; ++c) sum += __expf(z[c] - m);
    float lse = m + logf(sum);
    #pragma unroll
    for (int c = 0; c < N_CLS; ++c) out[(size_t)n * N_CLS + c] = z[c] - lse;
}

extern "C" void kernel_launch(void* const* d_in, const int* in_sizes, int n_in,
                              void* d_out, int out_size, void* d_ws, size_t ws_size,
                              hipStream_t stream) {
    const float* x    = (const float*)d_in[0];
    const int*   ei   = (const int*)  d_in[1];
    const float* W1   = (const float*)d_in[2];
    const float* b1   = (const float*)d_in[3];
    const float* beta = (const float*)d_in[4];
    const float* W2   = (const float*)d_in[5];
    const float* b2   = (const float*)d_in[6];
    float* out = (float*)d_out;

    const int N = in_sizes[0] / N_FEAT;     // 100000
    const int E = in_sizes[1] / 2;          // 3200000
    const int Etot = E + N;                 // + self loops

    float* ws = (float*)d_ws;
    unsigned* wpack = (unsigned*)ws;                 // 4096 dwords
    float* h   = ws + 4096;                          // N*16
    float* hn  = h  + (size_t)N * HID;               // N*16
    float* nrm = hn + (size_t)N * HID;               // N
    float* acc = nrm + N;                            // N*16
    float* den = acc + (size_t)N * HID;              // N (contiguous after acc)

    const int* srcA = ei;
    const int* dstA = ei + E;

    // 1) pack W1 into WMMA B-fragment layout
    agnn_pack_w1<<<1, 32, 0, stream>>>(W1, wpack);

    // 2) h = relu(x @ W1 + b1) via WMMA
    int nTiles = N / 16;
    agnn_gemm1<<<(nTiles + 7) / 8, 256, 0, stream>>>(x, wpack, b1, h, nTiles);

    // 3) two AGNN conv layers
    int nThN   = (N + 255) / 256;
    int nThZ   = (N * (HID + 1) + 255) / 256;
    int nThE   = ((Etot * 16) + 255) / 256;
    int nThD   = (N * HID + 255) / 256;
    for (int layer = 0; layer < 2; ++layer) {
        agnn_normalize<<<nThN, 256, 0, stream>>>(h, hn, nrm, N);
        agnn_zero<<<nThZ, 256, 0, stream>>>(acc, N * (HID + 1));   // acc + denom contiguous
        agnn_edges<<<nThE, 256, 0, stream>>>(srcA, dstA, E, Etot, hn, nrm, acc, den,
                                             beta, /*useBeta=*/layer);
        agnn_divide<<<nThD, 256, 0, stream>>>(acc, den, h, N * HID);
    }

    // 4) classifier head + log_softmax
    agnn_head<<<nThN, 256, 0, stream>>>(h, W2, b2, out, N);
}
